// InternLM3SelfAttention_15083925144098
// MI455X (gfx1250) — compile-verified, matrix operand order checked by reading, not run on hardware
//
#include <hip/hip_runtime.h>
#include <hip/hip_bf16.h>

// ---------------- problem constants ----------------
#define TT   2048
#define HH   4096
#define NH   32
#define NKV  8
#define HD   128
#define QKVW 6144          // (NH + 2*NKV) * HD
#define HALF 64            // HD/2 for RoPE
#define L2F  0.31143075889f // log2(1e6)/64

typedef __attribute__((ext_vector_type(16))) __bf16 v16bf;
typedef __attribute__((ext_vector_type(8)))  __bf16 v8bf;
typedef __attribute__((ext_vector_type(8)))  float  v8f;
typedef __attribute__((ext_vector_type(4)))  unsigned int v4u;
typedef __attribute__((ext_vector_type(8)))  int v8i;
typedef __attribute__((ext_vector_type(4)))  int v4i;

// round-to-nearest-even f32 -> bf16
__device__ __forceinline__ __bf16 f2bf(float f) {
  union { float f; unsigned u; } in; in.f = f;
  unsigned u = in.u + 0x7FFFu + ((in.u >> 16) & 1u);
  unsigned short h = (unsigned short)(u >> 16);
  union { unsigned short s; __bf16 b; } out; out.s = h;
  return out.b;
}
__device__ __forceinline__ __bf16 to_bf(float f)  { return f2bf(f); }
__device__ __forceinline__ __bf16 to_bf(__bf16 b) { return b; }

// Load a 16-element bf16 A/B fragment for v_wmma_f32_16x16x32_bf16.
// rowp points at 32 contiguous K-values for this lane's row/col.
// Lanes 0-15: elems 0-7 = K0..7, elems 8-15 = K16..23
// Lanes 16-31: elems 0-7 = K8..15, elems 8-15 = K24..31   (ISA 7.12.2)
__device__ __forceinline__ v16bf load_frag(const __bf16* rowp, int khalf) {
  v8bf lo = *(const v8bf*)(rowp + khalf * 8);
  v8bf hi = *(const v8bf*)(rowp + 16 + khalf * 8);
  v16bf r;
#pragma unroll
  for (int i = 0; i < 8; ++i) { r[i] = lo[i]; r[i + 8] = hi[i]; }
  return r;
}

// ---------------- Tensor Data Mover: 2D tile -> LDS (ISA ch.8 D# layout) -------
// lds_off      : byte offset of destination inside workgroup LDS
// gaddr        : byte address of tile start in global memory
// tile_d0/d1   : tile extents (elements, 2B each); d0 is the contiguous dim
// tensor_d0/d1 : tensor extents (elements) for OOB handling
// stride0      : elements between consecutive dim-1 lines
// padi/pada    : D# pad_interval / pad_amount codes (LDS row padding)
__device__ __forceinline__ void tdm_load_2d(unsigned lds_off, const void* gaddr,
                                            unsigned tile_d0, unsigned tile_d1,
                                            unsigned tensor_d0, unsigned tensor_d1,
                                            unsigned long long stride0,
                                            unsigned padi, unsigned pada) {
  unsigned long long ga = (unsigned long long)(uintptr_t)gaddr;
  v4u g0;
  g0[0] = 1u;                                            // count=1, user descriptor
  g0[1] = lds_off;                                       // lds_addr [63:32]
  g0[2] = (unsigned)(ga & 0xFFFFFFFFu);                  // global_addr lo
  g0[3] = (unsigned)((ga >> 32) & 0x01FFFFFFu)
        | (2u << 30);                                    // type=2 ("image")
  v8i g1;
  g1[0] = (int)((1u << 16)                               // data_size = 2 bytes
              | (1u << 20)                               // pad_enable
              | (padi << 22) | (pada << 25));
  g1[1] = (int)((tensor_d0 & 0xFFFFu) << 16);            // tensor_dim0[15:0]
  g1[2] = (int)(((tensor_d0 >> 16) & 0xFFFFu)
              | ((tensor_d1 & 0xFFFFu) << 16));          // dim0 hi | dim1 lo
  g1[3] = (int)(((tensor_d1 >> 16) & 0xFFFFu)
              | ((tile_d0 & 0xFFFFu) << 16));            // dim1 hi | tile_dim0
  g1[4] = (int)(tile_d1 & 0xFFFFu);                      // tile_dim1; tile_dim2=0
  g1[5] = (int)(unsigned)(stride0 & 0xFFFFFFFFu);        // dim0_stride lo
  g1[6] = (int)(unsigned)((stride0 >> 32) & 0xFFFFu);    // dim0_stride hi
  g1[7] = 0;                                             // dim1_stride unused (2D)
  v4i z4 = {0, 0, 0, 0};
#if __clang_major__ >= 23
  v8i z8 = {0, 0, 0, 0, 0, 0, 0, 0};
  __builtin_amdgcn_tensor_load_to_lds(g0, g1, z4, z4, z8, 0);
#else
  __builtin_amdgcn_tensor_load_to_lds(g0, g1, z4, z4, 0);
#endif
}

// ---------------- tiled WMMA GEMM: C[M,N] f32 = A[M,K] x B[K,N] (bf16 compute) --
#define BM 128
#define BN 128
#define BK 32
#define LPAD 40   // padded LDS row stride (elems); 80B rows keep 16B alignment

template <typename TA, typename TB>
__global__ __launch_bounds__(256)
void gemm_wmma(const TA* __restrict__ A, const TB* __restrict__ B,
               float* __restrict__ C, int M, int N, int K) {
  __shared__ __align__(16) __bf16 sA[BM][LPAD];   // [m][k]
  __shared__ __align__(16) __bf16 sB[BN][LPAD];   // transposed: [n][k]

  const int tid   = threadIdx.x;
  const int lane  = tid & 31;
  const int wid   = tid >> 5;
  const int l16   = lane & 15;
  const int khalf = lane >> 4;
  const int waveM = wid >> 1;                 // 4 wave-rows x 32
  const int waveN = wid & 1;                  // 2 wave-cols x 64
  const int blockRow = blockIdx.y * BM;
  const int blockCol = blockIdx.x * BN;

  v8f acc[2][4] = {};

  const int arow = tid >> 1;                  // 0..127
  const int acol = (tid & 1) * 16;            // 0 / 16
  const int bk   = tid >> 3;                  // 0..31
  const int bn0  = (tid & 7) * 16;            // 0..112

  for (int kt = 0; kt < K; kt += BK) {
    __syncthreads();
    const TA* ap = A + (size_t)(blockRow + arow) * K + kt + acol;
#pragma unroll
    for (int i = 0; i < 16; ++i) sA[arow][acol + i] = to_bf(ap[i]);
    const TB* bp = B + (size_t)(kt + bk) * N + blockCol + bn0;
#pragma unroll
    for (int i = 0; i < 16; ++i) sB[bn0 + i][bk] = to_bf(bp[i]);
    // GL2 prefetch of next K tile (speculative; dropped if OOB)
    __builtin_prefetch(ap + BK, 0, 1);
    __builtin_prefetch(bp + (size_t)BK * N, 0, 1);
    __syncthreads();

    v16bf af[2], bfg[4];
#pragma unroll
    for (int tm = 0; tm < 2; ++tm)
      af[tm] = load_frag(&sA[waveM * 32 + tm * 16 + l16][0], khalf);
#pragma unroll
    for (int tn = 0; tn < 4; ++tn)
      bfg[tn] = load_frag(&sB[waveN * 64 + tn * 16 + l16][0], khalf);
#pragma unroll
    for (int tm = 0; tm < 2; ++tm)
#pragma unroll
      for (int tn = 0; tn < 4; ++tn)
        acc[tm][tn] = __builtin_amdgcn_wmma_f32_16x16x32_bf16(
            false, af[tm], false, bfg[tn], (short)0, acc[tm][tn], false, false);
  }

#pragma unroll
  for (int tm = 0; tm < 2; ++tm) {
    const int row0 = blockRow + waveM * 32 + tm * 16 + khalf * 8;
#pragma unroll
    for (int tn = 0; tn < 4; ++tn) {
      const int col = blockCol + waveN * 64 + tn * 16 + l16;
#pragma unroll
      for (int r = 0; r < 8; ++r)
        C[(size_t)(row0 + r) * N + col] = acc[tm][tn][r];
    }
  }
}

// ---------------- RoPE + repack for attention ----------------
// Qh [NH][T][HD], Kt [NKV][T][HD], Vt [NKV][HD][T]  (all bf16)
__global__ __launch_bounds__(256)
void rope_pack(const float* __restrict__ qkv, const int* __restrict__ pos,
               __bf16* __restrict__ Qh, __bf16* __restrict__ Kt,
               __bf16* __restrict__ Vt) {
  const int t = blockIdx.x;
  const float p = (float)pos[t];
  const float* row = qkv + (size_t)t * QKVW;

  for (int i = threadIdx.x; i < NH * HALF; i += blockDim.x) {
    const int h = i / HALF, j = i % HALF;
    const float fr = p * exp2f(-(float)j * L2F);
    const float c = __cosf(fr), s = __sinf(fr);
    const float x0 = row[h * HD + j];
    const float x1 = row[h * HD + j + HALF];
    __bf16* q = Qh + ((size_t)h * TT + t) * HD;
    q[j]        = f2bf(x0 * c - x1 * s);
    q[j + HALF] = f2bf(x1 * c + x0 * s);
  }
  for (int i = threadIdx.x; i < NKV * HALF; i += blockDim.x) {
    const int kv = i / HALF, j = i % HALF;
    const float fr = p * exp2f(-(float)j * L2F);
    const float c = __cosf(fr), s = __sinf(fr);
    const float x0 = row[NH * HD + kv * HD + j];
    const float x1 = row[NH * HD + kv * HD + j + HALF];
    __bf16* k = Kt + ((size_t)kv * TT + t) * HD;
    k[j]        = f2bf(x0 * c - x1 * s);
    k[j + HALF] = f2bf(x1 * c + x0 * s);
  }
  for (int i = threadIdx.x; i < NKV * HD; i += blockDim.x) {
    const int kv = i / HD, d = i % HD;
    Vt[((size_t)kv * HD + d) * TT + t] = f2bf(row[(NH + NKV) * HD + i]);
  }
}

// ---------------- flash attention with TDM-staged K/V chunks ----------------
// One head x 128 query rows per WG; 8 waves of 16 rows each.
// K chunk in LDS: 32 keys x 128 hd, row stride 136 elems (TDM pad: 64dw+4dw)
// V chunk in LDS: 128 hd  x 32 keys, row stride 40 elems (TDM pad: 16dw+4dw)
#define SKLD 136
#define SVLD 40
__global__ __launch_bounds__(256)
void attn_flash(const __bf16* __restrict__ Qh, const __bf16* __restrict__ Kt,
                const __bf16* __restrict__ Vt, __bf16* __restrict__ Ob) {
  __shared__ __align__(16) __bf16 sK[32 * SKLD + 16];
  __shared__ __align__(16) __bf16 sV[128 * SVLD + 16];
  __shared__ __align__(16) __bf16 pLds[8][16][32];  // per-wave P tile

  const int h   = blockIdx.x;
  const int kv  = h >> 2;                    // NH/NKV = 4
  const int tid = threadIdx.x;
  const int lane = tid & 31, wid = tid >> 5;
  const int l16 = lane & 15, khalf = lane >> 4;
  const int blockRow = blockIdx.y * 128;
  const int qrow0 = blockRow + wid * 16;

  // Q fragments (16 rows x 128 K) held for the whole loop
  v16bf qf[4];
  const __bf16* qrowp = Qh + ((size_t)h * TT + qrow0 + l16) * HD;
#pragma unroll
  for (int c = 0; c < 4; ++c) qf[c] = load_frag(qrowp + 32 * c, khalf);

  v8f o[8] = {};
  float m[8], l[8];
#pragma unroll
  for (int r = 0; r < 8; ++r) { m[r] = -1e30f; l[r] = 0.f; }

  const float scale = 0.08838834764831845f;  // 1/sqrt(128)
  const unsigned skOff = (unsigned)(uintptr_t)&sK[0];
  const unsigned svOff = (unsigned)(uintptr_t)&sV[0];
  const int nch = (blockRow + 128) >> 5;     // block-uniform chunk count

  for (int kc = 0; kc < nch; ++kc) {
    const int k0 = kc * 32;
    __syncthreads();                         // all waves done with previous tiles
    if (wid == 0) {
      // K chunk: 32 rows of 128 contiguous bf16, stride HD
      tdm_load_2d(skOff, Kt + ((size_t)kv * TT + k0) * HD,
                  /*tile*/ HD, 32, /*tensor*/ HD, TT,
                  /*stride0*/ HD, /*padi 64dw*/ 5, /*pada 4dw*/ 3);
      // V chunk: 128 rows of 32 contiguous bf16 (token dim), stride T
      tdm_load_2d(svOff, Vt + (size_t)kv * HD * TT + k0,
                  /*tile*/ 32, HD, /*tensor*/ TT, HD,
                  /*stride0*/ TT, /*padi 16dw*/ 3, /*pada 4dw*/ 3);
      __builtin_amdgcn_s_wait_tensorcnt(0);
    }
    __syncthreads();                         // tiles visible to all waves

    if (k0 <= qrow0 + 15) {                  // causal: chunk intersects this wave
      v8f s[2] = {};
#pragma unroll
      for (int t = 0; t < 2; ++t) {
        const __bf16* krowp = sK + (t * 16 + l16) * SKLD;
#pragma unroll
        for (int c = 0; c < 4; ++c) {
          v16bf kb = load_frag(krowp + 32 * c, khalf);
          s[t] = __builtin_amdgcn_wmma_f32_16x16x32_bf16(
              false, qf[c], false, kb, (short)0, s[t], false, false);
        }
      }

      float corr[8];
#pragma unroll
      for (int r = 0; r < 8; ++r) {
        const int rg = qrow0 + khalf * 8 + r;
        float v0 = s[0][r] * scale;
        float v1 = s[1][r] * scale;
        if (k0 + l16 > rg)      v0 = -1e30f;
        if (k0 + 16 + l16 > rg) v1 = -1e30f;
        float mx = fmaxf(v0, v1);
#pragma unroll
        for (int off = 1; off < 16; off <<= 1)
          mx = fmaxf(mx, __shfl_xor(mx, off, 16));
        const float mnew = fmaxf(m[r], mx);
        corr[r] = __expf(m[r] - mnew);
        m[r] = mnew;
        const float p0 = __expf(v0 - mnew);
        const float p1 = __expf(v1 - mnew);
        float ps = p0 + p1;
#pragma unroll
        for (int off = 1; off < 16; off <<= 1)
          ps += __shfl_xor(ps, off, 16);
        l[r] = l[r] * corr[r] + ps;
        pLds[wid][khalf * 8 + r][l16]      = f2bf(p0);  // same-wave DS: in-order
        pLds[wid][khalf * 8 + r][16 + l16] = f2bf(p1);
      }

#pragma unroll
      for (int n = 0; n < 8; ++n)
#pragma unroll
        for (int r = 0; r < 8; ++r) o[n][r] *= corr[r];

      v16bf pf = load_frag(&pLds[wid][l16][0], khalf);
#pragma unroll
      for (int n = 0; n < 8; ++n) {
        v16bf vb = load_frag(sV + (n * 16 + l16) * SVLD, khalf);
        o[n] = __builtin_amdgcn_wmma_f32_16x16x32_bf16(
            false, pf, false, vb, (short)0, o[n], false, false);
      }
    }
  }

#pragma unroll
  for (int n = 0; n < 8; ++n) {
    const int col = h * HD + n * 16 + l16;
#pragma unroll
    for (int r = 0; r < 8; ++r) {
      const int rg = qrow0 + khalf * 8 + r;
      Ob[(size_t)rg * (NH * HD) + col] = f2bf(o[n][r] / l[r]);
    }
  }
}

// ---------------- host launch ----------------
extern "C" void kernel_launch(void* const* d_in, const int* in_sizes, int n_in,
                              void* d_out, int out_size, void* d_ws, size_t ws_size,
                              hipStream_t stream) {
  const int*   positions = (const int*)d_in[0];
  const float* hidden    = (const float*)d_in[1];
  const float* wqkv      = (const float*)d_in[2];
  const float* wo        = (const float*)d_in[3];
  float*       out       = (float*)d_out;

  char* ws = (char*)d_ws;
  size_t off = 0;
  float*  qkv   = (float*)(ws + off);  off += (size_t)TT * QKVW * sizeof(float); // 48 MB
  __bf16* Qh    = (__bf16*)(ws + off); off += (size_t)NH  * TT * HD * 2;         // 16 MB
  __bf16* Kt    = (__bf16*)(ws + off); off += (size_t)NKV * TT * HD * 2;         //  4 MB
  __bf16* Vt    = (__bf16*)(ws + off); off += (size_t)NKV * HD * TT * 2;         //  4 MB
  __bf16* attnb = (__bf16*)(ws + off); off += (size_t)TT  * HH * 2;              // 16 MB

  dim3 blk(256);
  // 1) fused QKV projection
  gemm_wmma<float, float><<<dim3(QKVW / BN, TT / BM), blk, 0, stream>>>(
      hidden, wqkv, qkv, TT, QKVW, HH);
  // 2) RoPE + repack Q/K/V
  rope_pack<<<dim3(TT), blk, 0, stream>>>(qkv, positions, Qh, Kt, Vt);
  // 3) causal flash attention (GQA via kv = h/4), TDM-staged K/V
  attn_flash<<<dim3(NH, TT / 128), blk, 0, stream>>>(Qh, Kt, Vt, attnb);
  // 4) output projection
  gemm_wmma<__bf16, float><<<dim3(HH / BN, TT / BM), blk, 0, stream>>>(
      attnb, wo, out, TT, HH, HH);
}
